// ConvFeatureGet0_47064251629663
// MI455X (gfx1250) — matrix operand order repaired and checked by reading, not compiled, not run on hardware
//
#include <hip/hip_runtime.h>
#include <stdint.h>

// ---------------- problem constants ----------------
#define SEQ     1024
#define DCH     128
#define OUTLEN  512
#define NBATCH  256
#define NCOLS   179          // 128 + 17*3
#define XROWS   608          // padded rows per batch in xbuf (valid: 0..585)
#define XVALID  586          // x rows actually needed: t in [438, 1023]
#define XBASE   438
#define JJPAD   544          // 532 extended positions padded to 17*32
#define NCHAN   51           // 17 convs * 3 channels (padded to 64 = 4 N-tiles)
#define TOTTAPS 110          // 13 + 55 + 21 + 21
#define LROWS   86           // LDS window rows: 32 positions + 54 max extra taps

// ws layout (bytes)
#define XBUF_BYTES  ((size_t)NBATCH * XROWS * DCH * 2)          // 39,845,888
#define WBUF_BYTES  ((size_t)TOTTAPS * 4 * 512 * 2)             //    450,560
#define YBUF_OFF    (XBUF_BYTES + WBUF_BYTES)                   // 40,296,448

typedef __attribute__((ext_vector_type(16))) __bf16 v16bf;
typedef __attribute__((ext_vector_type(8)))  float  v8f;

struct Ptrs17 { const float* p[17]; };

__device__ __forceinline__ unsigned short f32_to_bf16(float f) {
    unsigned int u = __float_as_uint(f);
    u += 0x7fffu + ((u >> 16) & 1u);     // round-to-nearest-even
    return (unsigned short)(u >> 16);
}

// kernel widths for the 17 convs: base1..9, max1..4, ave1..4
__device__ __constant__ int c_KS[17] = {1,2,3,5,8,13,21,34,55, 5,8,13,21, 5,8,13,21};

// -------------------------------------------------------------------------
// 1) convert x slice to bf16, swizzled into the WMMA 16-bit A-matrix order:
//    per 32-channel chunk store channels [0..7, 16..23, 8..15, 24..31]
//    so lane<16 reads its 16 A elements {K0..7,K16..23} contiguously and
//    lane>=16 reads {K8..15,K24..31} contiguously.
// -------------------------------------------------------------------------
__global__ __launch_bounds__(256) void convert_x_kernel(
    const float* __restrict__ x, unsigned short* __restrict__ xbuf)
{
    unsigned int idx = blockIdx.x * 256u + threadIdx.x;
    if (idx >= (unsigned)NBATCH * XROWS * DCH) return;
    int d = idx % DCH;
    int u = (idx / DCH) % XROWS;
    int b = idx / (DCH * XROWS);
    int chunk = d >> 5, d32 = d & 31;
    int grp = (d32 >> 3) & 3;                          // 0..3 within chunk
    int s = (grp == 1 || grp == 2) ? (d32 ^ 24) : d32; // swap middle groups
    unsigned short v = 0;
    if (u < XVALID) {
        float f = x[(size_t)b * SEQ * DCH + (size_t)(XBASE + u) * DCH + d];
        v = f32_to_bf16(f);
    }
    xbuf[((size_t)b * XROWS + u) * DCH + chunk * 32 + s] = v;
}

// -------------------------------------------------------------------------
// 2) pack reversed-tap, zero-padded bf16 weights in B-matrix lane layout:
//    wbuf[((woff+m)*4 + chunk)*512 + lane*16 + e] =
//        Wrev[ch = tile*16 + lane%16][m][d = chunk*32 + (lane/16)*8 + e%8 + (e/8)*16]
//    tile tap counts {13,55,21,21}, offsets {0,13,68,89}.
// -------------------------------------------------------------------------
__global__ __launch_bounds__(256) void pack_w_kernel(
    Ptrs17 W, unsigned short* __restrict__ wbuf)
{
    int idx = blockIdx.x * 256 + threadIdx.x;
    if (idx >= TOTTAPS * 4 * 512) return;
    int e     =  idx        & 15;
    int lane  = (idx >> 4)  & 31;
    int chunk = (idx >> 9)  & 3;
    int g     =  idx >> 11;                 // 0..109 global tap row
    int tile, m;
    if      (g < 13) { tile = 0; m = g; }
    else if (g < 68) { tile = 1; m = g - 13; }
    else if (g < 89) { tile = 2; m = g - 68; }
    else             { tile = 3; m = g - 89; }
    int c = tile * 16 + (lane & 15);        // padded channel id 0..63
    int d = chunk * 32 + ((lane >> 4) << 3) + (e & 7) + ((e >> 3) << 4);
    unsigned short v = 0;
    if (c < NCHAN) {
        int cv = c / 3, o = c % 3, k = c_KS[cv];
        if (m < k) {
            // w shape (3,1,k,128); reversed tap: k-1-m
            float f = W.p[cv][(size_t)o * k * DCH + (size_t)(k - 1 - m) * DCH + d];
            v = f32_to_bf16(f);
        }
    }
    wbuf[idx] = v;
}

// -------------------------------------------------------------------------
// 3) WMMA GEMM: one 128-thread block = 4 waves = 4 channel tiles sharing an
//    LDS-staged x window (contiguous 86x128 bf16 slab, 22 KB).
//    Each wave: acc[2 x 16x16] += A(pos x 32ch, from LDS) * B(32ch x 16oc).
// -------------------------------------------------------------------------
__global__ __launch_bounds__(128) void conv_wmma_kernel(
    const unsigned short* __restrict__ xbuf,
    const unsigned short* __restrict__ wbuf,
    float* __restrict__ ybuf)
{
    __shared__ __align__(16) unsigned short s_x[LROWS * DCH];   // 22,016 B

    const int tid  = threadIdx.x;
    const int lane = tid & 31;
    const int tile = tid >> 5;              // wave id = channel tile 0..3
    const int jp   = blockIdx.x;            // 0..16  (two 16-pos tiles each)
    const int b    = blockIdx.y;            // 0..255
    const int jj0  = jp * 32;
    const unsigned short* xb = xbuf + (size_t)b * XROWS * DCH;

    // Cooperative stage: rows u = jj0 .. jj0+85 are contiguous in xbuf.
    {
        const uint4* src = reinterpret_cast<const uint4*>(xb + (size_t)jj0 * DCH);
        uint4* dst = reinterpret_cast<uint4*>(s_x);
        for (int i = tid; i < (LROWS * DCH) / 8; i += 128)   // 1376 uint4
            dst[i] = src[i];
    }
    __syncthreads();

    const int taps = (tile == 0) ? 13 : (tile == 1) ? 55 : 21;
    const int woff = (tile == 0) ? 0  : (tile == 1) ? 13 : (tile == 2) ? 68 : 89;
    const int mrow = lane & 15;
    const int ksel = (lane >> 4) << 4;      // 0 or 16 (A-layout K half)

    v8f acc0 = {0.f,0.f,0.f,0.f,0.f,0.f,0.f,0.f};
    v8f acc1 = {0.f,0.f,0.f,0.f,0.f,0.f,0.f,0.f};

    for (int m = 0; m < taps; ++m) {
        // position jj uses x row u = 54 + jj - m; LDS row = u - jj0
        const int r0 = 54 + mrow - m;                      // 0..69
        const unsigned short* arow0 = s_x + r0 * DCH + ksel;
        const unsigned short* arow1 = arow0 + 16 * DCH;    // rows up to 85
        const unsigned short* wrow  = wbuf + (size_t)(woff + m) * 4 * 512 + lane * 16;
        #pragma unroll
        for (int ch = 0; ch < 4; ++ch) {
            v16bf bm = *reinterpret_cast<const v16bf*>(wrow  + ch * 512);
            v16bf a0 = *reinterpret_cast<const v16bf*>(arow0 + ch * 32);
            v16bf a1 = *reinterpret_cast<const v16bf*>(arow1 + ch * 32);
            acc0 = __builtin_amdgcn_wmma_f32_16x16x32_bf16(
                       false, a0, false, bm, (short)0, acc0, false, false);
            acc1 = __builtin_amdgcn_wmma_f32_16x16x32_bf16(
                       false, a1, false, bm, (short)0, acc1, false, false);
        }
    }

    // C/D layout: VGPR r, lane l -> M = r + 8*(l>>4), N = l&15.
    // r = 0..7 are 8 consecutive jj positions -> one v8f (2x b128) store each.
    const int n     = lane & 15;
    const int mbase = (lane >> 4) << 3;
    float* yrow = ybuf + ((size_t)b * 64 + (tile * 16 + n)) * JJPAD;
    *reinterpret_cast<v8f*>(yrow + jj0 + mbase)      = acc0;
    *reinterpret_cast<v8f*>(yrow + jj0 + 16 + mbase) = acc1;
}

// -------------------------------------------------------------------------
// 4) finalize: x passthrough (exact f32), bias add, max/avg pool, concat.
//    Bias after pooling is exact (constant shift for max, linear for avg).
// -------------------------------------------------------------------------
__global__ __launch_bounds__(256) void finalize_kernel(
    const float* __restrict__ x, const float* __restrict__ ybuf,
    Ptrs17 Bp, float* __restrict__ out)
{
    const int j = blockIdx.x;       // 0..511
    const int b = blockIdx.y;       // 0..255
    const int col = threadIdx.x;
    if (col >= NCOLS) return;
    float* orow = out + ((size_t)b * OUTLEN + j) * NCOLS;
    if (col < DCH) {
        orow[col] = x[(size_t)b * SEQ * DCH + (size_t)(OUTLEN + j) * DCH + col];
        return;
    }
    const int c  = col - DCH;       // 0..50
    const int jj = j + 20;
    const float* yr = ybuf + ((size_t)b * 64 + c) * JJPAD;
    const int cv = c / 3, o = c % 3;
    float val;
    if (c < 27) {                               // base branches
        val = yr[jj];
    } else if (c < 39) {                        // max pool branches
        const int p = c_KS[cv];                 // cv 9..12 -> 5,8,13,21
        float mx = yr[jj];
        for (int q = 1; q < p; ++q) mx = fmaxf(mx, yr[jj - q]);
        val = mx;
    } else {                                    // avg pool branches
        const int p = c_KS[cv];                 // cv 13..16 -> 5,8,13,21
        float s = 0.f;
        for (int q = 0; q < p; ++q) s += yr[jj - q];
        val = s / (float)p;
    }
    orow[col] = val + Bp.p[cv][o];
}

// -------------------------------------------------------------------------
extern "C" void kernel_launch(void* const* d_in, const int* in_sizes, int n_in,
                              void* d_out, int out_size, void* d_ws, size_t ws_size,
                              hipStream_t stream)
{
    (void)in_sizes; (void)n_in; (void)out_size; (void)ws_size;
    const float* x = (const float*)d_in[0];

    // setup_inputs() dict order interleaves w/b per conv:
    // 0:x, {1,2}:w/b_base1 .. {17,18}:w/b_base9,
    // {19,20}:w/b_max1 .. {25,26}:w/b_max4, {27,28}:w/b_ave1 .. {33,34}:w/b_ave4
    Ptrs17 W, Bp;
    for (int i = 0; i < 9; ++i) { W.p[i]      = (const float*)d_in[1  + 2*i];
                                  Bp.p[i]     = (const float*)d_in[2  + 2*i]; }
    for (int i = 0; i < 4; ++i) { W.p[9 + i]  = (const float*)d_in[19 + 2*i];
                                  Bp.p[9 + i] = (const float*)d_in[20 + 2*i]; }
    for (int i = 0; i < 4; ++i) { W.p[13 + i] = (const float*)d_in[27 + 2*i];
                                  Bp.p[13 + i]= (const float*)d_in[28 + 2*i]; }

    char* ws = (char*)d_ws;
    unsigned short* xbuf = (unsigned short*)ws;
    unsigned short* wbuf = (unsigned short*)(ws + XBUF_BYTES);
    float*          ybuf = (float*)(ws + YBUF_OFF);

    {   // 1) x -> bf16 swizzled slice
        unsigned int n = (unsigned)NBATCH * XROWS * DCH;
        convert_x_kernel<<<(n + 255u) / 256u, 256, 0, stream>>>(x, xbuf);
    }
    {   // 2) weights -> bf16 packed B matrices
        int n = TOTTAPS * 4 * 512;
        pack_w_kernel<<<(n + 255) / 256, 256, 0, stream>>>(W, wbuf);
    }
    {   // 3) all 17 convs as WMMA tap-sum GEMM (4 tiles share LDS window)
        conv_wmma_kernel<<<dim3(17, NBATCH), 128, 0, stream>>>(xbuf, wbuf, ybuf);
    }
    {   // 4) pooling + bias + concat
        finalize_kernel<<<dim3(OUTLEN, NBATCH), 256, 0, stream>>>(x, ybuf, Bp, (float*)d_out);
    }
}
